// SymmetrizedResNet_54090818126211
// MI455X (gfx1250) — matrix operand order repaired and checked by reading, not compiled, not run on hardware
//
#include <hip/hip_runtime.h>

typedef _Float16 f16x16 __attribute__((ext_vector_type(16)));
typedef _Float16 f16x8  __attribute__((ext_vector_type(8)));
typedef float    f32x8  __attribute__((ext_vector_type(8)));

struct NetParams {
  const float *j, *jet_scaler, *dijet_scaler, *quadjet_scaler;
  const float *jetPtGBN, *jetEtaGBN, *jetMassGBN;
  const float *jetEmbed_W, *jetEmbed_b;
  const float *dijetGBN, *dijetEmbed1_W, *dijetEmbed1_b;
  const float *quadjetGBN, *quadjetEmbed_W, *quadjetEmbed_b;
  const float *dR1_W, *dR1_b, *dR1_GBN;
  const float *dconvJ_W, *dconvJ_b;
  const float *dR2_W, *dR2_b;
  const float *dijetEmbed2_W, *dijetEmbed2_b;
  const float *qR1_W, *qR1_b, *qR1_GBN;
  const float *qconvD_W, *qconvD_b;
  const float *qR2_W, *qR2_b;
  const float *eventConv1_W, *eventConv1_b, *eventConv1_GBN;
  const float *eventConv2_W, *eventConv2_b;
  const float *select_q_W, *select_q_b, *select_q_GBN;
  const float *out_W, *out_b, *out_GBN;
  float* out;
  int n;
};

__device__ __forceinline__ float frcp(float x) { return __builtin_amdgcn_rcpf(x); }
__device__ __forceinline__ float frsq(float x) { return __builtin_amdgcn_rsqf(x); }
// silu via fast rcp: x * 1/(1+exp(-x))
__device__ __forceinline__ float siluf(float x) { return x * frcp(1.0f + __expf(-x)); }

__device__ __forceinline__ float wrapf(float d) {
  const float PI = 3.14159265358979323846f;
  const float TWO_PI = 6.28318530717958647692f;
  float x = d + PI;
  x = x - TWO_PI * floorf(x * (1.0f / TWO_PI));
  return x - PI;
}

__device__ __forceinline__ float gbn1(float x, const float* g, int C, int c, bool bias) {
  float y = __ldg(g + c) * (x - __ldg(g + 2 * C + c)) * frsq(__ldg(g + 3 * C + c) + 1e-5f);
  return bias ? y + __ldg(g + C + c) : y;
}

// ---- Fused epilogue constants: y = mul*x + add (per 16-row block, 8 ch per lane) ----
struct Epi { f32x8 mul[2], add[2]; };

__device__ __forceinline__ Epi makeLinear(const float* bias, int lane) {
  Epi e;
#pragma unroll
  for (int blk = 0; blk < 2; ++blk) {
    int cb = (blk << 4) + ((lane >> 4) << 3);
#pragma unroll
    for (int r = 0; r < 8; ++r) { e.mul[blk][r] = 1.0f; e.add[blk][r] = __ldg(bias + cb + r); }
  }
  return e;
}

// GBN folded: y = s*x + (s*(b-mu)+beta), s = g*rsqrt(var+eps)
__device__ __forceinline__ Epi makeGBN(const float* bias, const float* G, int lane) {
  Epi e;
#pragma unroll
  for (int blk = 0; blk < 2; ++blk) {
    int cb = (blk << 4) + ((lane >> 4) << 3);
#pragma unroll
    for (int r = 0; r < 8; ++r) {
      int c = cb + r;
      float s = __ldg(G + c) * frsq(__ldg(G + 96 + c) + 1e-5f);
      e.mul[blk][r] = s;
      e.add[blk][r] = s * (__ldg(bias + c) - __ldg(G + 64 + c)) + __ldg(G + 32 + c);
    }
  }
  return e;
}

// ---- WMMA fragment helpers (wave32; A and B both use contiguous k = 16*(lane/16) + s) ----

// B fragment: region layout [event(16)][channel(32)] f16 at element offset `base`.
__device__ __forceinline__ f16x16 loadB(const _Float16* act, int base, int lane) {
  int o = base + ((lane & 15) << 5) + ((lane >> 4) << 4);
  f16x8 lo = *(const f16x8*)(act + o);
  f16x8 hi = *(const f16x8*)(act + o + 8);
  f16x16 b;
#pragma unroll
  for (int i = 0; i < 8; ++i) { b[i] = lo[i]; b[i + 8] = hi[i]; }
  return b;
}

// A fragment from f32 weights in global memory, stride in floats between K elements.
__device__ __forceinline__ f16x16 packA(const float* W, int blk, int stride, int lane) {
  const float* row = W + (blk * 16 + (lane & 15)) * 32 * stride;
  int kb = (lane >> 4) << 4;
  f16x16 a;
#pragma unroll
  for (int i = 0; i < 8; ++i) {
    a[i]     = (_Float16)__ldg(row + (kb + i) * stride);
    a[i + 8] = (_Float16)__ldg(row + (kb + 8 + i) * stride);
  }
  return a;
}

// A fragment for K-padded embed weights: W is (32 x F) row-major, K>=F reads as 0.
template <int F>
__device__ __forceinline__ f16x16 packA_pad(const float* W, int blk, int lane) {
  const float* row = W + (blk * 16 + (lane & 15)) * F;
  int kb = (lane >> 4) << 4;
  f16x16 a;
#pragma unroll
  for (int i = 0; i < 8; ++i) {
    int k0 = kb + i, k1 = kb + 8 + i;
    a[i]     = (k0 < F) ? (_Float16)__ldg(row + k0) : (_Float16)0.0f;
    a[i + 8] = (k1 < F) ? (_Float16)__ldg(row + k1) : (_Float16)0.0f;
  }
  return a;
}

__device__ __forceinline__ f32x8 wmma16(f16x16 a, f16x16 b, f32x8 c) {
  return __builtin_amdgcn_wmma_f32_16x16x32_f16(false, a, false, b, (short)0, c,
                                                false, false);
}

// D store + fused epilogue. SR=false: y = mul*x+add ; SR=true: y = silu(mul*x+add+res)
template <bool SR>
__device__ __forceinline__ void storeD(_Float16* act, int dBase, const _Float16* res,
                                       int rBase, const Epi& e, int blk, f32x8 acc,
                                       int lane) {
  int cb = (blk << 4) + ((lane >> 4) << 3);
  int off = ((lane & 15) << 5) + cb;
  f16x8 o;
  f16x8 rv;
  if (SR) rv = *(const f16x8*)(res + rBase + off);
#pragma unroll
  for (int r = 0; r < 8; ++r) {
    float x = e.mul[blk][r] * acc[r] + e.add[blk][r];
    if (SR) x = siluf(x + (float)rv[r]);
    o[r] = (_Float16)x;
  }
  *(f16x8*)(act + dBase + off) = o;
}

// Dual store for embeds: raw = x+b (residual), act = silu(x+b)
__device__ __forceinline__ void storeDual(_Float16* raw, _Float16* act, int base,
                                          const Epi& e, int blk, f32x8 acc, int lane) {
  int cb = (blk << 4) + ((lane >> 4) << 3);
  int off = base + ((lane & 15) << 5) + cb;
  f16x8 o0, o1;
#pragma unroll
  for (int r = 0; r < 8; ++r) {
    float x = acc[r] + e.add[blk][r];
    o0[r] = (_Float16)x;
    o1[r] = (_Float16)siluf(x);
  }
  *(f16x8*)(raw + off) = o0;
  *(f16x8*)(act + off) = o1;
}

// Embed layer via zero-padded-K WMMA: raw holds padded input, becomes pre-activation.
template <int F>
__device__ __forceinline__ void embedLayer(_Float16* raw, _Float16* act, const float* W,
                                           const float* bias, int Pn, int lane) {
  f16x16 A0 = packA_pad<F>(W, 0, lane), A1 = packA_pad<F>(W, 1, lane);
  Epi e = makeLinear(bias, lane);
#pragma unroll 2
  for (int pp = 0; pp < Pn; ++pp) {
    f16x16 b = loadB(raw, pp * 512, lane);
    f32x8 acc0 = {}, acc1 = {};
    acc0 = wmma16(A0, b, acc0);
    acc1 = wmma16(A1, b, acc1);
    storeDual(raw, act, pp * 512, e, 0, acc0, lane);
    storeDual(raw, act, pp * 512, e, 1, acc1, lane);
  }
}

// reinforce: out[o,k] = sum_c,d W[o,c,d]*{main[c,2k], main[c,2k+1], anc[c,k]} + epi
template <bool SR>
__device__ __forceinline__ void reinforceLayer(const _Float16* mainBuf, _Float16* ancBuf,
                                               const _Float16* resBuf, const float* W3,
                                               const Epi& e, int K, int lane) {
  f16x16 A00 = packA(W3 + 0, 0, 3, lane), A01 = packA(W3 + 1, 0, 3, lane),
         A02 = packA(W3 + 2, 0, 3, lane);
  f16x16 A10 = packA(W3 + 0, 1, 3, lane), A11 = packA(W3 + 1, 1, 3, lane),
         A12 = packA(W3 + 2, 1, 3, lane);
#pragma unroll 2
  for (int k = 0; k < K; ++k) {
    f16x16 b0 = loadB(mainBuf, (2 * k) * 512, lane);
    f16x16 b1 = loadB(mainBuf, (2 * k + 1) * 512, lane);
    f16x16 b2 = loadB(ancBuf, k * 512, lane);
    f32x8 acc0 = {};
    acc0 = wmma16(A00, b0, acc0);
    acc0 = wmma16(A01, b1, acc0);
    acc0 = wmma16(A02, b2, acc0);
    f32x8 acc1 = {};
    acc1 = wmma16(A10, b0, acc1);
    acc1 = wmma16(A11, b1, acc1);
    acc1 = wmma16(A12, b2, acc1);
    storeD<SR>(ancBuf, k * 512, resBuf, k * 512, e, 0, acc0, lane);
    storeD<SR>(ancBuf, k * 512, resBuf, k * 512, e, 1, acc1, lane);
  }
}

// in-place 32x32 conv over Pn positions
template <bool SR>
__device__ __forceinline__ void convLayer(_Float16* buf, const _Float16* resBuf,
                                          const float* W, const Epi& e, int Pn,
                                          int lane) {
  f16x16 A0 = packA(W, 0, 1, lane), A1 = packA(W, 1, 1, lane);
#pragma unroll 2
  for (int pp = 0; pp < Pn; ++pp) {
    f16x16 b = loadB(buf, pp * 512, lane);
    f32x8 acc0 = {}, acc1 = {};
    acc0 = wmma16(A0, b, acc0);
    acc1 = wmma16(A1, b, acc1);
    storeD<SR>(buf, pp * 512, resBuf, pp * 512, e, 0, acc0, lane);
    storeD<SR>(buf, pp * 512, resBuf, pp * 512, e, 1, acc1, lane);
  }
}

__device__ __forceinline__ void siluResPass(_Float16* x, const _Float16* x0, int count,
                                            int lane) {
  for (int i = lane; i < count; i += 32)
    x[i] = (_Float16)siluf((float)x[i] + (float)x0[i]);
}

__global__ __launch_bounds__(32, 4) void srn_kernel(NetParams p) {
  alignas(16) __shared__ _Float16 sJJ[12 * 512];
  alignas(16) __shared__ _Float16 sJ0[12 * 512];
  alignas(16) __shared__ _Float16 sD[6 * 512];
  alignas(16) __shared__ _Float16 sD0[6 * 512];
  alignas(16) __shared__ _Float16 sQ[3 * 512];
  alignas(16) __shared__ _Float16 sQ0[3 * 512];
  __shared__ float sS[48];

  int lane = threadIdx.x;
  int tile = blockIdx.x;

  // ---------- zero-fill padded embed-input buffers ----------
  {
    f16x8 z = {};
    for (int i = lane * 8; i < 12 * 512; i += 32 * 8) *(f16x8*)(sJ0 + i) = z;
    for (int i = lane * 8; i < 6 * 512; i += 32 * 8) *(f16x8*)(sD0 + i) = z;
    for (int i = lane * 8; i < 3 * 512; i += 32 * 8) *(f16x8*)(sQ0 + i) = z;
  }
  __syncthreads();

  // ---------- Stage A: ancillary features (one event per lane, lanes 0-15) ----------
  if (lane < 16) {
    const float4* j4 = (const float4*)(p.j + (tile * 16 + lane) * 16);
    float4 f0 = __ldg(j4 + 0), f1 = __ldg(j4 + 1), f2 = __ldg(j4 + 2), f3 = __ldg(j4 + 3);
    float pt[4] = {f0.x, f0.y, f0.z, f0.w};
    float eta[4] = {f1.x, f1.y, f1.z, f1.w};
    float phi[4] = {f2.x, f2.y, f2.z, f2.w};
    float mm[4] = {f3.x, f3.y, f3.z, f3.w};
    float px[4], py[4], pz[4], En[4];
    float sE = 0.f, sx = 0.f, sy = 0.f, sz = 0.f;
#pragma unroll
    for (int i = 0; i < 4; ++i) {
      px[i] = pt[i] * cosf(phi[i]);
      py[i] = pt[i] * sinf(phi[i]);
      pz[i] = pt[i] * sinhf(eta[i]);
      En[i] = sqrtf(px[i] * px[i] + py[i] * py[i] + pz[i] * pz[i] + mm[i] * mm[i]);
      sE += En[i]; sx += px[i]; sy += py[i]; sz += pz[i];
    }
    const int DJA[6] = {0, 2, 0, 1, 0, 1}, DJB[6] = {1, 3, 2, 3, 3, 2};
    float etad[6], phid[6];
    const float* dsc = p.dijet_scaler;
    float rd0 = frcp(__ldg(dsc + 2)), rd1 = frcp(__ldg(dsc + 3));
#pragma unroll
    for (int k = 0; k < 6; ++k) {
      int a = DJA[k], b = DJB[k];
      float Px = px[a] + px[b], Py = py[a] + py[b], Pz = pz[a] + pz[b];
      float Ed = En[a] + En[b];
      float pt2 = Px * Px + Py * Py;
      float ptd = sqrtf(pt2);
      float md = sqrtf(fmaxf(Ed * Ed - pt2 - Pz * Pz, 1e-8f));
      float de = eta[a] - eta[b];
      float dp = wrapf(phi[a] - phi[b]);
      float dR = sqrtf(de * de + dp * dp);
      sD0[k * 512 + lane * 32 + 0] =
          (_Float16)gbn1((md - __ldg(dsc + 0)) * rd0, p.dijetGBN, 2, 0, true);
      sD0[k * 512 + lane * 32 + 1] =
          (_Float16)gbn1((dR - __ldg(dsc + 1)) * rd1, p.dijetGBN, 2, 1, true);
      etad[k] = asinhf(Pz * frcp(fmaxf(ptd, 1e-8f)));
      phid[k] = atan2f(Py, Px);
    }
    float m4 = sqrtf(fmaxf(sE * sE - sx * sx - sy * sy - sz * sz, 1e-8f));
    const int QJA[3] = {0, 2, 4}, QJB[3] = {1, 3, 5};
    const float* qsc = p.quadjet_scaler;
    float rq0 = frcp(__ldg(qsc + 2)), rq1 = frcp(__ldg(qsc + 3));
#pragma unroll
    for (int t = 0; t < 3; ++t) {
      int a = QJA[t], b = QJB[t];
      float de = etad[a] - etad[b], dp = wrapf(phid[a] - phid[b]);
      float dRdd = sqrtf(de * de + dp * dp);
      sQ0[t * 512 + lane * 32 + 0] =
          (_Float16)gbn1((dRdd - __ldg(qsc + 0)) * rq0, p.quadjetGBN, 2, 0, true);
      sQ0[t * 512 + lane * 32 + 1] =
          (_Float16)gbn1((m4 - __ldg(qsc + 1)) * rq1, p.quadjetGBN, 2, 1, true);
    }
    const int JREP[12] = {0, 1, 2, 3, 0, 2, 1, 3, 0, 3, 1, 2};
    const float* jsc = p.jet_scaler;
    float rj0 = frcp(__ldg(jsc + 4)), rj1 = frcp(__ldg(jsc + 5));
    float rj2 = frcp(__ldg(jsc + 6)), rj3 = frcp(__ldg(jsc + 7));
#pragma unroll
    for (int pp = 0; pp < 12; ++pp) {
      int jj = JREP[pp];
      int o = pp * 512 + lane * 32;
      sJ0[o + 0] = (_Float16)gbn1((pt[jj] - __ldg(jsc + 0)) * rj0, p.jetPtGBN, 1, 0, true);
      sJ0[o + 1] = (_Float16)gbn1((eta[jj] - __ldg(jsc + 1)) * rj1, p.jetEtaGBN, 1, 0, false);
      sJ0[o + 2] = (_Float16)((phi[jj] - __ldg(jsc + 2)) * rj2);
      sJ0[o + 3] = (_Float16)gbn1((mm[jj] - __ldg(jsc + 3)) * rj3, p.jetMassGBN, 1, 0, true);
    }
  }
  __syncthreads();

  // ---------- Stage B: embeds via zero-padded-K WMMA ----------
  embedLayer<4>(sJ0, sJJ, p.jetEmbed_W, p.jetEmbed_b, 12, lane);
  embedLayer<2>(sD0, sD, p.dijetEmbed1_W, p.dijetEmbed1_b, 6, lane);
  embedLayer<2>(sQ0, sQ, p.quadjetEmbed_W, p.quadjetEmbed_b, 3, lane);
  __syncthreads();

  // ---------- dijet block ----------
  {
    Epi e = makeGBN(p.dR1_b, p.dR1_GBN, lane);     // d = gbn(reinforce(jj,d)+b)
    reinforceLayer<false>(sJJ, sD, sD0, p.dR1_W, e, 6, lane);
  }
  __syncthreads();
  {
    Epi e = makeLinear(p.dconvJ_b, lane);          // jj = silu(c1(jj)+b+j0)
    convLayer<true>(sJJ, sJ0, p.dconvJ_W, e, 12, lane);
  }
  __syncthreads();
  siluResPass(sD, sD0, 6 * 512, lane);             // d = silu(d+d0)
  __syncthreads();
  {
    Epi e = makeLinear(p.dR2_b, lane);             // d = silu(reinforce(jj,d)+b+d0)
    reinforceLayer<true>(sJJ, sD, sD0, p.dR2_W, e, 6, lane);
  }
  __syncthreads();
  {
    Epi e = makeLinear(p.dijetEmbed2_b, lane);     // d = silu(c1(d)+b+d0)
    convLayer<true>(sD, sD0, p.dijetEmbed2_W, e, 6, lane);
  }
  __syncthreads();

  // ---------- quadjet block ----------
  {
    Epi e = makeGBN(p.qR1_b, p.qR1_GBN, lane);     // q = gbn(reinforce(d,q)+b)
    reinforceLayer<false>(sD, sQ, sQ0, p.qR1_W, e, 3, lane);
  }
  __syncthreads();
  {
    Epi e = makeLinear(p.qconvD_b, lane);          // d = silu(c1(d)+b+d0)
    convLayer<true>(sD, sD0, p.qconvD_W, e, 6, lane);
  }
  __syncthreads();
  siluResPass(sQ, sQ0, 3 * 512, lane);             // q = silu(q+q0)
  __syncthreads();
  {
    Epi e = makeLinear(p.qR2_b, lane);             // q = silu(reinforce(d,q)+b+q0)
    reinforceLayer<true>(sD, sQ, sQ0, p.qR2_W, e, 3, lane);
  }
  __syncthreads();
  {
    Epi e = makeGBN(p.eventConv1_b, p.eventConv1_GBN, lane); // q = silu(gbn(c1(q)+b)+q0)
    convLayer<true>(sQ, sQ0, p.eventConv1_W, e, 3, lane);
  }
  __syncthreads();
  {
    Epi e = makeLinear(p.eventConv2_b, lane);      // q = silu(c1(q)+b+q0)
    convLayer<true>(sQ, sQ0, p.eventConv2_W, e, 3, lane);
  }
  __syncthreads();

  // ---------- select / softmax / output head ----------
  for (int i = lane; i < 48; i += 32) {
    int e = i & 15, pp = i >> 4;
    float acc = __ldg(p.select_q_b);
#pragma unroll
    for (int c = 0; c < 32; ++c)
      acc += __ldg(p.select_q_W + c) * (float)sQ[pp * 512 + e * 32 + c];
    sS[pp * 16 + e] = gbn1(acc, p.select_q_GBN, 1, 0, true);
  }
  __syncthreads();

  if (lane < 16) {
    float s0 = sS[lane], s1 = sS[16 + lane], s2 = sS[32 + lane];
    float mx = fmaxf(s0, fmaxf(s1, s2));
    float e0 = __expf(s0 - mx), e1 = __expf(s1 - mx), e2 = __expf(s2 - mx);
    float inv = frcp(e0 + e1 + e2);
    float q0s = e0 * inv, q1s = e1 * inv, q2s = e2 * inv;
    float acc = __ldg(p.out_b);
#pragma unroll
    for (int c = 0; c < 32; ++c) {
      float ev = q0s * (float)sQ[0 * 512 + lane * 32 + c] +
                 q1s * (float)sQ[1 * 512 + lane * 32 + c] +
                 q2s * (float)sQ[2 * 512 + lane * 32 + c];
      acc += __ldg(p.out_W + c) * ev;
    }
    acc = gbn1(acc, p.out_GBN, 1, 0, true);
    int n0 = tile * 16 + lane;
    p.out[n0] = acc;                  // c_score (n,1)
    p.out[p.n + n0 * 3 + 0] = q0s;    // q_score (n,3)
    p.out[p.n + n0 * 3 + 1] = q1s;
    p.out[p.n + n0 * 3 + 2] = q2s;
  }
}

extern "C" void kernel_launch(void* const* d_in, const int* in_sizes, int n_in,
                              void* d_out, int out_size, void* d_ws, size_t ws_size,
                              hipStream_t stream) {
  (void)n_in; (void)out_size; (void)d_ws; (void)ws_size;
  NetParams p;
  p.j = (const float*)d_in[0];
  p.jet_scaler = (const float*)d_in[1];
  p.dijet_scaler = (const float*)d_in[2];
  p.quadjet_scaler = (const float*)d_in[3];
  p.jetPtGBN = (const float*)d_in[4];
  p.jetEtaGBN = (const float*)d_in[5];
  p.jetMassGBN = (const float*)d_in[6];
  p.jetEmbed_W = (const float*)d_in[7];
  p.jetEmbed_b = (const float*)d_in[8];
  p.dijetGBN = (const float*)d_in[9];
  p.dijetEmbed1_W = (const float*)d_in[10];
  p.dijetEmbed1_b = (const float*)d_in[11];
  p.quadjetGBN = (const float*)d_in[12];
  p.quadjetEmbed_W = (const float*)d_in[13];
  p.quadjetEmbed_b = (const float*)d_in[14];
  p.dR1_W = (const float*)d_in[15];
  p.dR1_b = (const float*)d_in[16];
  p.dR1_GBN = (const float*)d_in[17];
  p.dconvJ_W = (const float*)d_in[18];
  p.dconvJ_b = (const float*)d_in[19];
  p.dR2_W = (const float*)d_in[20];
  p.dR2_b = (const float*)d_in[21];
  p.dijetEmbed2_W = (const float*)d_in[22];
  p.dijetEmbed2_b = (const float*)d_in[23];
  p.qR1_W = (const float*)d_in[24];
  p.qR1_b = (const float*)d_in[25];
  p.qR1_GBN = (const float*)d_in[26];
  p.qconvD_W = (const float*)d_in[27];
  p.qconvD_b = (const float*)d_in[28];
  p.qR2_W = (const float*)d_in[29];
  p.qR2_b = (const float*)d_in[30];
  p.eventConv1_W = (const float*)d_in[31];
  p.eventConv1_b = (const float*)d_in[32];
  p.eventConv1_GBN = (const float*)d_in[33];
  p.eventConv2_W = (const float*)d_in[34];
  p.eventConv2_b = (const float*)d_in[35];
  p.select_q_W = (const float*)d_in[36];
  p.select_q_b = (const float*)d_in[37];
  p.select_q_GBN = (const float*)d_in[38];
  p.out_W = (const float*)d_in[39];
  p.out_b = (const float*)d_in[40];
  p.out_GBN = (const float*)d_in[41];
  p.out = (float*)d_out;
  p.n = in_sizes[0] / 16;
  int tiles = p.n / 16;
  hipLaunchKernelGGL(srn_kernel, dim3(tiles), dim3(32), 0, stream, p);
}
